// BTC_LSTM_60370060313246
// MI455X (gfx1250) — compile-verified, matrix operand order checked by reading, not run on hardware
//
#include <hip/hip_runtime.h>

// LSTM B=4096, S=512, I=1, H=50 (gates=200), PyTorch order i,f,g,o.
// One workgroup = 16 batch rows, 13 wave32. W_hh^T is staged gate-interleaved:
// B column n = 16t + 4*jj + g  <-  W_hh row g*50 + (4t + jj).
// So wave t's 16x16 WMMA tile holds all 4 gates of hidden indices 4t..4t+3,
// making the gate stage a same-wave LDS exchange (no barrier). h is double-
// buffered -> exactly ONE workgroup barrier per timestep. Cell state c lives
// in registers; W_ih/bias constants are hoisted into registers pre-loop.

#define BATCH   4096
#define SEQ     512
#define HID     50
#define BT      16            // batch rows per workgroup (M)
#define NT      13            // N tiles of 16 (200 -> 208)
#define HS      72            // h row stride (halves), padded vs bank conflicts
#define WS      72            // Whh row stride (halves)
#define ZR      20            // per-wave z-slice row stride (floats), padded
#define NTHREADS (NT * 32)    // 416 threads

typedef __attribute__((ext_vector_type(16))) _Float16 v16h;
typedef __attribute__((ext_vector_type(8)))  _Float16 v8h;
typedef __attribute__((ext_vector_type(2)))  _Float16 v2h;
typedef __attribute__((ext_vector_type(8)))  float    v8f;
typedef __attribute__((ext_vector_type(4)))  float    v4f;

__device__ __forceinline__ float fast_sigmoid(float x) {
    return __builtin_amdgcn_rcpf(1.0f + __expf(-x));
}
__device__ __forceinline__ float fast_tanh(float x) {
    float e = __expf(2.0f * x);
    return 1.0f - 2.0f * __builtin_amdgcn_rcpf(e + 1.0f);
}

__global__ __launch_bounds__(NTHREADS, 1)
void BTC_LSTM_60370060313246_kernel(const float* __restrict__ x,
                                    const float* __restrict__ W_ih,
                                    const float* __restrict__ W_hh,
                                    const float* __restrict__ b_ih,
                                    const float* __restrict__ b_hh,
                                    const float* __restrict__ W_fc,
                                    const float* __restrict__ b_fc,
                                    float* __restrict__ out) {
    __shared__ __align__(16) _Float16 sWhh[208 * WS];   // gate-interleaved W_hh^T, f16
    __shared__ __align__(16) _Float16 sH[2 * BT * HS];  // double-buffered h, f16
    __shared__ __align__(16) float    sZ[NT * BT * ZR]; // per-wave private D slices

    const int tid = threadIdx.x;
    const int b0  = blockIdx.x * BT;

    // ---------------- one-time staging ----------------
    {
        uint32_t* wz = (uint32_t*)sWhh;
        for (int i = tid; i < 208 * WS / 2; i += NTHREADS) wz[i] = 0u;
        uint32_t* hz = (uint32_t*)sH;
        for (int i = tid; i < 2 * BT * HS / 2; i += NTHREADS) hz[i] = 0u;
    }
    __syncthreads();
    // gate-interleaved staging: column n -> (t=n>>4, jj=(n>>2)&3, g=n&3), j=4t+jj
    for (int i = tid; i < 208 * HID; i += NTHREADS) {
        int n = i / HID;
        int k = i - n * HID;
        int g  = n & 3;
        int jj = (n >> 2) & 3;
        int j  = 4 * (n >> 4) + jj;
        if (j < HID)
            sWhh[n * WS + k] = (_Float16)W_hh[(g * HID + j) * HID + k];
    }
    __syncthreads();

    const int lane = tid & 31;
    const int wave = tid >> 5;
    const int lr   = lane & 15;
    const int lh   = lane >> 4;
    const int n0   = wave * 16;

    // persistent per-wave B fragments (held in VGPRs for all 512 steps)
    v16h bfrag[2];
    #pragma unroll
    for (int kt = 0; kt < 2; ++kt) {
        v8h lo = *(const v8h*)&sWhh[(n0 + lr) * WS + kt * 32 + lh * 8];
        v8h hi = *(const v8h*)&sWhh[(n0 + lr) * WS + kt * 32 + 16 + lh * 8];
        #pragma unroll
        for (int i = 0; i < 8; ++i) { bfrag[kt][i] = lo[i]; bfrag[kt][i + 8] = hi[i]; }
    }

    // elementwise ownership: lane handles (row r, hidden j0) and (r, j0+1)
    const int  r   = lr;
    const int  jj0 = lh * 2;
    const int  j0  = wave * 4 + jj0;
    const bool act = (j0 < HID);           // wave-12 lanes 16-31 are padding

    float wih_c[2][4], bias_c[2][4];       // hoisted constants (registers)
    #pragma unroll
    for (int p = 0; p < 2; ++p) {
        #pragma unroll
        for (int g = 0; g < 4; ++g) {
            int idx = g * HID + j0 + p;
            wih_c[p][g]  = act ? W_ih[idx] : 0.0f;
            bias_c[p][g] = act ? (b_ih[idx] + b_hh[idx]) : 0.0f;
        }
    }

    const float* xrow = x + (size_t)(b0 + r) * SEQ;
    float* zs = sZ + wave * (BT * ZR);     // this wave's private D slice
    float c0 = 0.0f, c1 = 0.0f;            // cell state in registers

    // ---------------- recurrence: ONE barrier per step ----------------
    for (int s = 0; s < SEQ; ++s) {
        if ((s & 63) == 0) {               // uniform condition: no exec juggling
            int ps = (s + 64 < SEQ) ? s + 64 : SEQ - 1;
            __builtin_prefetch(xrow + ps, 0, 3);
        }
        float xv = xrow[s];                // issued early, hidden behind WMMA

        const _Float16* hb = (const _Float16*)sH + (s & 1) * (BT * HS);
        v8h lo0 = *(const v8h*)&hb[lr * HS +      lh * 8];
        v8h hi0 = *(const v8h*)&hb[lr * HS + 16 + lh * 8];
        v8h lo1 = *(const v8h*)&hb[lr * HS + 32 + lh * 8];
        v8h hi1 = *(const v8h*)&hb[lr * HS + 48 + lh * 8];
        v16h a0, a1;
        #pragma unroll
        for (int i = 0; i < 8; ++i) {
            a0[i] = lo0[i]; a0[i + 8] = hi0[i];
            a1[i] = lo1[i]; a1[i + 8] = hi1[i];
        }
        v8f acc = {};
        acc = __builtin_amdgcn_wmma_f32_16x16x32_f16(false, a0, false, bfrag[0],
                                                     (short)0, acc, false, false);
        acc = __builtin_amdgcn_wmma_f32_16x16x32_f16(false, a1, false, bfrag[1],
                                                     (short)0, acc, false, false);
        #pragma unroll
        for (int v = 0; v < 8; ++v)
            zs[(lh * 8 + v) * ZR + lr] = acc[v];

        // gates: same-wave DS ordering makes own slice readable w/o barrier
        _Float16* hw = (_Float16*)sH + ((s + 1) & 1) * (BT * HS);
        if (act) {
            v4f z0 = *(const v4f*)&zs[r * ZR + 4 * jj0];        // i,f,g,o of (r,j0)
            v4f z1 = *(const v4f*)&zs[r * ZR + 4 * (jj0 + 1)];  // i,f,g,o of (r,j0+1)
            float zi0 = z0[0] + xv * wih_c[0][0] + bias_c[0][0];
            float zf0 = z0[1] + xv * wih_c[0][1] + bias_c[0][1];
            float zg0 = z0[2] + xv * wih_c[0][2] + bias_c[0][2];
            float zo0 = z0[3] + xv * wih_c[0][3] + bias_c[0][3];
            float zi1 = z1[0] + xv * wih_c[1][0] + bias_c[1][0];
            float zf1 = z1[1] + xv * wih_c[1][1] + bias_c[1][1];
            float zg1 = z1[2] + xv * wih_c[1][2] + bias_c[1][2];
            float zo1 = z1[3] + xv * wih_c[1][3] + bias_c[1][3];
            c0 = fast_sigmoid(zf0) * c0 + fast_sigmoid(zi0) * fast_tanh(zg0);
            c1 = fast_sigmoid(zf1) * c1 + fast_sigmoid(zi1) * fast_tanh(zg1);
            float h0 = fast_sigmoid(zo0) * fast_tanh(c0);
            float h1 = fast_sigmoid(zo1) * fast_tanh(c1);
            v2h hv; hv[0] = (_Float16)h0; hv[1] = (_Float16)h1;
            *(v2h*)&hw[r * HS + j0] = hv;  // packed 4B store (j0 is even)
        }
        __syncthreads();                   // the single per-step barrier
    }

    // ---------------- FC head: out = sigmoid(h @ W_fc^T + b_fc) ----------------
    if (tid < BT) {
        const _Float16* hf = (const _Float16*)sH;  // SEQ even -> final h in buffer 0
        float a = b_fc[0];
        #pragma unroll 5
        for (int j = 0; j < HID; ++j)
            a += (float)hf[tid * HS + j] * W_fc[j];
        out[b0 + tid] = fast_sigmoid(a);
    }
}

extern "C" void kernel_launch(void* const* d_in, const int* in_sizes, int n_in,
                              void* d_out, int out_size, void* d_ws, size_t ws_size,
                              hipStream_t stream) {
    const float* x    = (const float*)d_in[0];
    const float* W_ih = (const float*)d_in[1];
    const float* W_hh = (const float*)d_in[2];
    const float* b_ih = (const float*)d_in[3];
    const float* b_hh = (const float*)d_in[4];
    const float* W_fc = (const float*)d_in[5];
    const float* b_fc = (const float*)d_in[6];
    float* out = (float*)d_out;

    dim3 grid(BATCH / BT);
    dim3 block(NTHREADS);
    BTC_LSTM_60370060313246_kernel<<<grid, block, 0, stream>>>(
        x, W_ih, W_hh, b_ih, b_hh, W_fc, b_fc, out);
}
